// DPCA3D_30477087932763
// MI455X (gfx1250) — matrix operand order repaired and checked by reading, not compiled, not run on hardware
//
#include <hip/hip_runtime.h>
#include <math.h>

// ---------------------------------------------------------------------------
// DPCA-3D for MI455X (gfx1250, wave32, WMMA).
// Compute-bound pipeline (~66 GFLOP, ~60MB working set << 192MB L2):
// all GEMM-like stages run on v_wmma_f32_16x16x32_f16 with f32 accumulation.
// GEMM staging uses GLOBAL_LOAD_ASYNC_TO_LDS (ASYNCcnt) when the builtin
// exists, else load->ds_store fallback.
// ---------------------------------------------------------------------------

#define HEADS    8
#define DH       64
#define TOPK     1024
#define KCENT    1024
#define KM_ITERS 4
#define BATCH    2
#define CDIM     256
#define INNER    512
#define SQ       4096   // 16*16*16 query spatial
#define SK       2048   // 8*16*16 context spatial
#define BHN      (BATCH*HEADS)  // 16

typedef _Float16 h16;
typedef __attribute__((ext_vector_type(16))) _Float16 v16h;
typedef __attribute__((ext_vector_type(8)))  float    v8f;

#if defined(__gfx1250__) && __has_builtin(__builtin_amdgcn_global_load_async_to_lds_b128) && __has_builtin(__builtin_amdgcn_s_wait_asynccnt)
#define USE_ASYNC_LDS 1
#else
#define USE_ASYNC_LDS 0
#endif

#if USE_ASYNC_LDS
// Param types per clang diagnostic: (AS1 int4*, AS3 int4*, imm offset, imm cpol)
typedef int v4i __attribute__((vector_size(4 * sizeof(int))));
typedef __attribute__((address_space(1))) v4i* gptr_v4i;
typedef __attribute__((address_space(3))) v4i* lptr_v4i;
__device__ __forceinline__ void async_copy_b128(const h16* g, h16* l) {
  __builtin_amdgcn_global_load_async_to_lds_b128((gptr_v4i)g, (lptr_v4i)l, 0, 0);
}
#endif

// ---------------- wave helpers (wave32) ----------------
__device__ __forceinline__ float wave_sum32(float v) {
#pragma unroll
  for (int m = 1; m < 32; m <<= 1) v += __shfl_xor(v, m, 32);
  return v;
}
__device__ __forceinline__ float grp16_sum(float v) {
#pragma unroll
  for (int m = 1; m < 16; m <<= 1) v += __shfl_xor(v, m, 32);
  return v;
}
__device__ __forceinline__ float grp16_max(float v) {
#pragma unroll
  for (int m = 1; m < 16; m <<= 1) v = fmaxf(v, __shfl_xor(v, m, 32));
  return v;
}

// ---------------- WMMA helpers ----------------
__device__ __forceinline__ v8f wmma_f16(v16h a, v16h b, v8f c) {
  // D = A(16x32 f16) * B(32x16 f16) + C(16x16 f32)
  return __builtin_amdgcn_wmma_f32_16x16x32_f16(false, a, false, b,
                                                (short)0, c, false, false);
}

// Gather an A fragment (16x32 f16) from a row-major [16][ld] tile.
// Layout: lane l holds row M=l%16; half h=l/16; element i -> K = 16h + i
//   (contiguous order, matching the B fragments)
__device__ __forceinline__ v16h frag_a(const h16* base, int ld) {
  const int l = threadIdx.x & 31;
  const int m = l & 15, hh = l >> 4;
  v16h f;
#pragma unroll
  for (int i = 0; i < 16; ++i) {
    const int k = 16 * hh + i;
    f[i] = base[m * ld + k];
  }
  return f;
}

// ---------------- elementwise / LN kernels ----------------
__global__ void k_f32_to_f16(const float* __restrict__ x, h16* __restrict__ y, int n) {
  int i = blockIdx.x * 256 + threadIdx.x;
  if (i < n) y[i] = (h16)x[i];
}

// chan_ln over channel axis; input x[b][c][p] (c-stride = S), output f16 [b*S+p][C]
__global__ void k_chan_ln(const float* __restrict__ x, const float* __restrict__ g,
                          const float* __restrict__ bt, h16* __restrict__ y, int S) {
  const int wid  = blockIdx.x * 8 + (threadIdx.x >> 5);
  const int lane = threadIdx.x & 31;
  const int b = wid / S, p = wid % S;
  const float* xb = x + (size_t)b * CDIM * S + p;
  float s = 0.f, s2 = 0.f;
#pragma unroll
  for (int c = lane; c < CDIM; c += 32) {
    float v = xb[(size_t)c * S];
    s += v; s2 += v * v;
  }
  s = wave_sum32(s); s2 = wave_sum32(s2);
  const float m   = s / (float)CDIM;
  const float var = s2 / (float)CDIM - m * m;
  const float inv = 1.0f / (sqrtf(fmaxf(var, 0.f)) + 1e-6f);
  h16* yb = y + (size_t)wid * CDIM;
#pragma unroll
  for (int c = lane; c < CDIM; c += 32) {
    float v = xb[(size_t)c * S];
    yb[c] = (h16)(g[c] * (v - m) * inv + bt[c]);
  }
}

// ---------------- generic WMMA GEMM:  C[M,N] = A[M,K] * B[N,K]^T ----------------
// A,B f16 row-major (K contiguous), C f32. Block tile 64x32, 8 waves (4x2 of 16x16),
// K-step 64 (two WMMAs per barrier). Staging via async-to-LDS when available.
__global__ void k_gemm_nt(const h16* __restrict__ A, const h16* __restrict__ B,
                          float* __restrict__ C, int M, int N, int K) {
  __shared__ __align__(32) h16 As[64 * 64];  // [m][k]
  __shared__ __align__(32) h16 Bs[32 * 64];  // [n][k]
  const int tid  = threadIdx.x;
  const int wave = tid >> 5, lane = tid & 31;
  const int wm = wave & 3, wn = wave >> 2;
  const int n16 = lane & 15, hh = lane >> 4;
  const int tilesN = N >> 5;
  const int bm = (blockIdx.x / tilesN) * 64;
  const int bn = (blockIdx.x % tilesN) * 32;
  v8f acc = {};
  const int ia = tid * 16, ra = ia >> 6, ca = ia & 63;  // A stage: 16 halves/thread
  const int ib = tid * 8,  rb = ib >> 6, cb = ib & 63;  // B stage:  8 halves/thread
  for (int k0 = 0; k0 < K; k0 += 64) {
    const h16* ga = &A[(size_t)(bm + ra) * K + k0 + ca];
    const h16* gb = &B[(size_t)(bn + rb) * K + k0 + cb];
    if (k0 + 64 < K) {  // L2 prefetch of next A slab -> global_prefetch_b8
      __builtin_prefetch(ga + 64, 0, 1);
    }
#if USE_ASYNC_LDS
    async_copy_b128(ga,     &As[ia]);
    async_copy_b128(ga + 8, &As[ia + 8]);
    async_copy_b128(gb,     &Bs[ib]);
    __builtin_amdgcn_s_wait_asynccnt(0);
#else
    *(int4*)&As[ia]     = *(const int4*)ga;
    *(int4*)&As[ia + 8] = *(const int4*)(ga + 8);
    *(int4*)&Bs[ib]     = *(const int4*)gb;
#endif
    __syncthreads();
    v16h a0 = frag_a(&As[wm * 16 * 64], 64);
    v16h a1 = frag_a(&As[wm * 16 * 64 + 32], 64);
    v16h b0 = *(const v16h*)&Bs[(wn * 16 + n16) * 64 + 16 * hh];
    v16h b1 = *(const v16h*)&Bs[(wn * 16 + n16) * 64 + 32 + 16 * hh];
    acc = wmma_f16(a0, b0, acc);
    acc = wmma_f16(a1, b1, acc);
    __syncthreads();
  }
  const int col = bn + wn * 16 + n16;
  const int row = bm + wm * 16 + 8 * hh;
#pragma unroll
  for (int r = 0; r < 8; ++r) C[(size_t)(row + r) * N + col] = acc[r];
}

// ---------------- fold heads (+ optional L2 normalize) ----------------
// out[(b*8+h)*S + n][0..63] = Y[(b*S+n)*ldY + colOff + h*64 + d] (/ ||.||)
__global__ void k_fold(const float* __restrict__ Y, h16* __restrict__ out,
                       int S, int ldY, int colOff, int doNorm) {
  const int w = blockIdx.x * 8 + (threadIdx.x >> 5);
  const int lane = threadIdx.x & 31;
  const int b = w / (8 * S);
  const int h = (w / S) & 7;
  const int n = w % S;
  const float* src = Y + ((size_t)(b * S + n)) * ldY + colOff + h * DH;
  const int d = lane * 2;
  const float v0 = src[d], v1 = src[d + 1];
  float inv = 1.0f;
  if (doNorm) {
    float s = wave_sum32(v0 * v0 + v1 * v1);
    inv = 1.0f / fmaxf(sqrtf(s), 1e-12f);
  }
  out[(size_t)w * DH + d]     = (h16)(v0 * inv);
  out[(size_t)w * DH + d + 1] = (h16)(v1 * inv);
}

// ---------------- k-means ----------------
__global__ void k_cent_init(const h16* __restrict__ qh, float* __restrict__ cent,
                            h16* __restrict__ centh) {
  int i = blockIdx.x * 256 + threadIdx.x;  // KCENT*DH
  float v = (float)qh[i];
  cent[i] = v; centh[i] = (h16)v;
}

__global__ void k_cnorm(const float* __restrict__ cent, float* __restrict__ cnorm) {
  const int w = blockIdx.x * 8 + (threadIdx.x >> 5);
  const int lane = threadIdx.x & 31;
  float s = 0.f;
#pragma unroll
  for (int d = lane; d < DH; d += 32) { float v = cent[w * DH + d]; s += v * v; }
  s = wave_sum32(s);
  if (lane == 0) cnorm[w] = s;
}

// Fused WMMA distance + running argmin over 1024 centroids.
// Block: 128 points, 8 waves of 16 rows. d2 = |x|^2 + (|c|^2 - 2 x.c); |x|^2 const per row.
__global__ void k_assign(const h16* __restrict__ X, const h16* __restrict__ Ch,
                         const float* __restrict__ cnorm, int* __restrict__ assign) {
  __shared__ __align__(32) h16 As[128 * 64];
  __shared__ __align__(32) h16 Bs[16 * 64];
  const int tid = threadIdx.x, wave = tid >> 5, lane = tid & 31;
  const int n16 = lane & 15, hh = lane >> 4;
  const int m0 = blockIdx.x * 128;
  {
    const int4* src = (const int4*)&X[(size_t)m0 * DH];
    int4* dst = (int4*)As;
#pragma unroll
    for (int i = 0; i < 4; ++i) dst[tid + 256 * i] = src[tid + 256 * i];
  }
  __syncthreads();
  const v16h a0 = frag_a(&As[wave * 16 * 64], 64);       // K 0..31
  const v16h a1 = frag_a(&As[wave * 16 * 64 + 32], 64);  // K 32..63
  float best[8]; int bidx[8];
#pragma unroll
  for (int r = 0; r < 8; ++r) { best[r] = 3.4e38f; bidx[r] = 0; }
  const int ibs = tid * 4;
  for (int nc = 0; nc < KCENT; nc += 16) {
    *(int2*)&Bs[ibs] = *(const int2*)&Ch[(size_t)(nc + (ibs >> 6)) * DH + (ibs & 63)];
    __syncthreads();
    v16h b0 = *(const v16h*)&Bs[n16 * 64 + 16 * hh];
    v16h b1 = *(const v16h*)&Bs[n16 * 64 + 32 + 16 * hh];
    v8f s = {};
    s = wmma_f16(a0, b0, s);
    s = wmma_f16(a1, b1, s);
    const float cn = cnorm[nc + n16];
    const int id = nc + n16;
#pragma unroll
    for (int r = 0; r < 8; ++r) {
      const float d2 = cn - 2.0f * s[r];
      if (d2 < best[r]) { best[r] = d2; bidx[r] = id; }
    }
    __syncthreads();
  }
#pragma unroll
  for (int r = 0; r < 8; ++r) {
#pragma unroll
    for (int msk = 1; msk < 16; msk <<= 1) {
      float ov = __shfl_xor(best[r], msk, 32);
      int   oi = __shfl_xor(bidx[r], msk, 32);
      if (ov < best[r] || (ov == best[r] && oi < bidx[r])) { best[r] = ov; bidx[r] = oi; }
    }
  }
  if (n16 == 0) {
    const int row = m0 + wave * 16 + 8 * hh;
#pragma unroll
    for (int r = 0; r < 8; ++r) assign[row + r] = bidx[r];
  }
}

__global__ void k_zero(float* __restrict__ p, int n) {
  int i = blockIdx.x * 256 + threadIdx.x;
  if (i < n) p[i] = 0.f;
}

__global__ void k_segsum(const h16* __restrict__ X, const int* __restrict__ assign,
                         float* __restrict__ sums, float* __restrict__ cnt) {
  const int w = blockIdx.x * 8 + (threadIdx.x >> 5);
  const int lane = threadIdx.x & 31;
  const int a = assign[w];
  const int d = lane * 2;
  atomicAdd(&sums[a * DH + d],     (float)X[(size_t)w * DH + d]);
  atomicAdd(&sums[a * DH + d + 1], (float)X[(size_t)w * DH + d + 1]);
  if (lane == 0) atomicAdd(&cnt[a], 1.0f);
}

__global__ void k_cent_update(const float* __restrict__ sums, const float* __restrict__ cnt,
                              float* __restrict__ cent, h16* __restrict__ centh) {
  int i = blockIdx.x * 256 + threadIdx.x;  // KCENT*DH
  const float n = cnt[i >> 6];
  const float v = (n > 0.f) ? sums[i] / fmaxf(n, 1.0f) : cent[i];
  cent[i] = v; centh[i] = (h16)v;
}

// ---------------- key scoring + exact stable top-k + gather ----------------
__global__ void k_dist(const h16* __restrict__ kh, const int* __restrict__ kc,
                       const float* __restrict__ cent, float* __restrict__ dist) {
  const int w = blockIdx.x * 8 + (threadIdx.x >> 5);
  const int lane = threadIdx.x & 31;
  const int a = kc[w];
  float s = 0.f;
#pragma unroll
  for (int d = lane; d < DH; d += 32)
    s += fabsf(cent[a * DH + d] - (float)kh[(size_t)w * DH + d]);
  s = wave_sum32(s);
  if (lane == 0) dist[w] = s;
}

// rank-by-counting == stable argsort(-dist)[:TOPK]; order fed to softmax is
// permutation-invariant, only the selected set (+ determinism) matters.
__global__ void k_topk(const float* __restrict__ dist, int* __restrict__ sel) {
  __shared__ float ds[SK];
  const int bh = blockIdx.x, tid = threadIdx.x;
  for (int e = tid; e < SK; e += 256) ds[e] = dist[bh * SK + e];
  __syncthreads();
  for (int e = tid; e < SK; e += 256) {
    const float de = ds[e];
    int rank = 0;
    for (int j = 0; j < SK; ++j) {
      const float dj = ds[j];
      rank += (dj > de) || (dj == de && j < e);
    }
    if (rank < TOPK) sel[bh * TOPK + rank] = e;
  }
}

__global__ void k_gather(const h16* __restrict__ kh, const h16* __restrict__ vh,
                         const int* __restrict__ sel, h16* __restrict__ ksel,
                         h16* __restrict__ vsel) {
  const int g = blockIdx.x * 256 + threadIdx.x;  // BHN*TOPK
  const int bh = g >> 10;
  const int src = sel[g];
  const int4* ks = (const int4*)&kh[((size_t)bh * SK + src) * DH];
  const int4* vs = (const int4*)&vh[((size_t)bh * SK + src) * DH];
  int4* kd = (int4*)&ksel[(size_t)g * DH];
  int4* vd = (int4*)&vsel[(size_t)g * DH];
#pragma unroll
  for (int i = 0; i < 8; ++i) { kd[i] = ks[i]; vd[i] = vs[i]; }
}

// ---------------- flash attention (wave = 16 query rows, 32-key tiles) ----------------
__global__ void k_attn(const h16* __restrict__ qh, const h16* __restrict__ ksel,
                       const h16* __restrict__ vsel, h16* __restrict__ attnH) {
  __shared__ __align__(32) h16 Ps[8 * 16 * 32];  // per-wave P tile (16x32)
  const int tid = threadIdx.x, wave = tid >> 5, lane = tid & 31;
  const int n16 = lane & 15, hh = lane >> 4;
  const int bh = blockIdx.x >> 5;      // 32 row-tiles per head-batch
  const int tile = blockIdx.x & 31;
  const int m0 = tile * 128 + wave * 16;
  const int b = bh >> 3, hsel = bh & 7;
  const h16* qb = qh + ((size_t)bh * SQ + m0) * DH;
  const v16h a0 = frag_a(qb, DH);
  const v16h a1 = frag_a(qb + 32, DH);
  const h16* kb = ksel + (size_t)bh * TOPK * DH;
  const h16* vb = vsel + (size_t)bh * TOPK * DH;
  h16* pw = &Ps[wave * 512];
  float mrow[8], lrow[8];
  v8f o0 = {}, o1 = {}, o2 = {}, o3 = {};
#pragma unroll
  for (int r = 0; r < 8; ++r) { mrow[r] = -3.4e38f; lrow[r] = 0.f; }
  for (int j0 = 0; j0 < TOPK; j0 += 32) {
    // S = q @ k_sel^T for 32 keys (two 16-col fragments, K=64 in two steps)
    const v16h bkA0 = *(const v16h*)&kb[(size_t)(j0 + n16) * DH + 16 * hh];
    const v16h bkA1 = *(const v16h*)&kb[(size_t)(j0 + n16) * DH + 32 + 16 * hh];
    const v16h bkB0 = *(const v16h*)&kb[(size_t)(j0 + 16 + n16) * DH + 16 * hh];
    const v16h bkB1 = *(const v16h*)&kb[(size_t)(j0 + 16 + n16) * DH + 32 + 16 * hh];
    v8f s0 = {}, s1 = {};
    s0 = wmma_f16(a0, bkA0, s0); s0 = wmma_f16(a1, bkA1, s0);
    s1 = wmma_f16(a0, bkB0, s1); s1 = wmma_f16(a1, bkB1, s1);
    // online softmax
#pragma unroll
    for (int r = 0; r < 8; ++r) {
      float tm = grp16_max(fmaxf(s0[r], s1[r]));
      const float nm = fmaxf(mrow[r], tm);
      const float sc = __expf(mrow[r] - nm);
      mrow[r] = nm;
      const float p0 = __expf(s0[r] - nm);
      const float p1 = __expf(s1[r] - nm);
      lrow[r] = lrow[r] * sc + grp16_sum(p0 + p1);
      pw[(r + 8 * hh) * 32 + n16]      = (h16)p0;
      pw[(r + 8 * hh) * 32 + 16 + n16] = (h16)p1;
      o0[r] *= sc; o1[r] *= sc; o2[r] *= sc; o3[r] *= sc;
    }
    __syncthreads();                       // C-frag layout -> A-frag layout via LDS
    const v16h pa = frag_a(pw, 32);
    __syncthreads();
    // O += P(16x32) @ V(32x64), four 16-col chunks of d
    v16h bv0, bv1, bv2, bv3;
#pragma unroll
    for (int i = 0; i < 16; ++i) {
      const h16* vrow = &vb[(size_t)(j0 + i + 16 * hh) * DH + n16];
      bv0[i] = vrow[0]; bv1[i] = vrow[16]; bv2[i] = vrow[32]; bv3[i] = vrow[48];
    }
    o0 = wmma_f16(pa, bv0, o0);
    o1 = wmma_f16(pa, bv1, o1);
    o2 = wmma_f16(pa, bv2, o2);
    o3 = wmma_f16(pa, bv3, o3);
  }
  // epilogue: normalize rows, write folded-back layout [b*SQ+p][h*64+d] (f16 for W_out GEMM)
  h16* ob = attnH + ((size_t)b * SQ + m0) * INNER + hsel * DH;
#pragma unroll
  for (int r = 0; r < 8; ++r) {
    const int m = r + 8 * hh;
    const float inv = 1.0f / lrow[r];
    h16* orow = ob + (size_t)m * INNER;
    orow[n16]      = (h16)(o0[r] * inv);
    orow[16 + n16] = (h16)(o1[r] * inv);
    orow[32 + n16] = (h16)(o2[r] * inv);
    orow[48 + n16] = (h16)(o3[r] * inv);
  }
}

// ---------------- final LN + gamma*out + residual ----------------
__global__ void k_final(const float* __restrict__ Y, const float* __restrict__ g,
                        const float* __restrict__ bt, const float* __restrict__ gamma,
                        const float* __restrict__ qsrc, float* __restrict__ out) {
  const int w = blockIdx.x * 8 + (threadIdx.x >> 5);  // (b,p)
  const int lane = threadIdx.x & 31;
  const int b = w >> 12, p = w & 4095;
  const float* y = Y + (size_t)w * CDIM;
  float s = 0.f, s2 = 0.f;
#pragma unroll
  for (int c = lane; c < CDIM; c += 32) { float v = y[c]; s += v; s2 += v * v; }
  s = wave_sum32(s); s2 = wave_sum32(s2);
  const float m   = s / (float)CDIM;
  const float var = s2 / (float)CDIM - m * m;
  const float inv = 1.0f / (sqrtf(fmaxf(var, 0.f)) + 1e-6f);
  const float gm = gamma[0];
#pragma unroll
  for (int c = lane; c < CDIM; c += 32) {
    const size_t oi = ((size_t)b * CDIM + c) * SQ + p;
    out[oi] = gm * (g[c] * (y[c] - m) * inv + bt[c]) + qsrc[oi];
  }
}

// ---------------------------------------------------------------------------
extern "C" void kernel_launch(void* const* d_in, const int* in_sizes, int n_in,
                              void* d_out, int out_size, void* d_ws, size_t ws_size,
                              hipStream_t stream) {
  (void)in_sizes; (void)n_in; (void)out_size; (void)ws_size;
  const float* qsrc = (const float*)d_in[0];
  const float* ctx  = (const float*)d_in[1];
  const float* w_q  = (const float*)d_in[2];
  const float* w_kv = (const float*)d_in[3];
  const float* w_out= (const float*)d_in[4];
  const float* cn_g = (const float*)d_in[5];
  const float* cn_b = (const float*)d_in[6];
  const float* qn_g = (const float*)d_in[7];
  const float* qn_b = (const float*)d_in[8];
  const float* on_g = (const float*)d_in[9];
  const float* on_b = (const float*)d_in[10];
  const float* gamma= (const float*)d_in[11];
  float* out = (float*)d_out;

  char* wsp = (char*)d_ws;
  auto alloc = [&](size_t bytes) -> void* {
    void* p = (void*)wsp;
    wsp += (bytes + 255) & ~(size_t)255;
    return p;
  };
  h16*   wqH   = (h16*)  alloc((size_t)INNER * CDIM * 2);
  h16*   wkvH  = (h16*)  alloc((size_t)2 * INNER * CDIM * 2);
  h16*   woutH = (h16*)  alloc((size_t)CDIM * INNER * 2);
  h16*   ctxn  = (h16*)  alloc((size_t)BATCH * SK * CDIM * 2);
  h16*   qsn   = (h16*)  alloc((size_t)BATCH * SQ * CDIM * 2);
  float* Ykv   = (float*)alloc((size_t)BATCH * SK * 2 * INNER * 4);
  float* Yq    = (float*)alloc((size_t)BATCH * SQ * INNER * 4);
  h16*   qh    = (h16*)  alloc((size_t)BHN * SQ * DH * 2);
  h16*   kh    = (h16*)  alloc((size_t)BHN * SK * DH * 2);
  h16*   vh    = (h16*)  alloc((size_t)BHN * SK * DH * 2);
  float* cent  = (float*)alloc((size_t)KCENT * DH * 4);
  h16*   centH = (h16*)  alloc((size_t)KCENT * DH * 2);
  float* cnorm = (float*)alloc((size_t)KCENT * 4);
  int*   asg   = (int*)  alloc((size_t)BHN * SQ * 4);
  float* sums  = (float*)alloc((size_t)KCENT * DH * 4);
  float* cnt   = (float*)alloc((size_t)KCENT * 4);
  int*   kc    = (int*)  alloc((size_t)BHN * SK * 4);
  float* dist  = (float*)alloc((size_t)BHN * SK * 4);
  int*   sel   = (int*)  alloc((size_t)BHN * TOPK * 4);
  h16*   ksel  = (h16*)  alloc((size_t)BHN * TOPK * DH * 2);
  h16*   vsel  = (h16*)  alloc((size_t)BHN * TOPK * DH * 2);
  h16*   attnH = (h16*)  alloc((size_t)BATCH * SQ * INNER * 2);
  float* Y3    = (float*)alloc((size_t)BATCH * SQ * CDIM * 4);

  const dim3 blk(256);
  // 1) weights -> f16
  k_f32_to_f16<<<dim3(INNER * CDIM / 256), blk, 0, stream>>>(w_q, wqH, INNER * CDIM);
  k_f32_to_f16<<<dim3(2 * INNER * CDIM / 256), blk, 0, stream>>>(w_kv, wkvH, 2 * INNER * CDIM);
  k_f32_to_f16<<<dim3(CDIM * INNER / 256), blk, 0, stream>>>(w_out, woutH, CDIM * INNER);
  // 2) channel LN -> f16 row-major [P][C]
  k_chan_ln<<<dim3(BATCH * SK / 8), blk, 0, stream>>>(ctx, cn_g, cn_b, ctxn, SK);
  k_chan_ln<<<dim3(BATCH * SQ / 8), blk, 0, stream>>>(qsrc, qn_g, qn_b, qsn, SQ);
  // 3) projections (WMMA GEMMs)
  k_gemm_nt<<<dim3((BATCH * SK / 64) * (2 * INNER / 32)), blk, 0, stream>>>(
      ctxn, wkvH, Ykv, BATCH * SK, 2 * INNER, CDIM);
  k_gemm_nt<<<dim3((BATCH * SQ / 64) * (INNER / 32)), blk, 0, stream>>>(
      qsn, wqH, Yq, BATCH * SQ, INNER, CDIM);
  // 4) fold heads, L2-normalize q/k
  k_fold<<<dim3(BHN * SQ / 8), blk, 0, stream>>>(Yq, qh, SQ, INNER, 0, 1);
  k_fold<<<dim3(BHN * SK / 8), blk, 0, stream>>>(Ykv, kh, SK, 2 * INNER, 0, 1);
  k_fold<<<dim3(BHN * SK / 8), blk, 0, stream>>>(Ykv, vh, SK, 2 * INNER, INNER, 0);
  // 5) k-means on normalized q points
  k_cent_init<<<dim3(KCENT * DH / 256), blk, 0, stream>>>(qh, cent, centH);
  k_cnorm<<<dim3(KCENT / 8), blk, 0, stream>>>(cent, cnorm);
  for (int it = 0; it < KM_ITERS; ++it) {
    k_assign<<<dim3(BHN * SQ / 128), blk, 0, stream>>>(qh, centH, cnorm, asg);
    k_zero<<<dim3(KCENT * DH / 256), blk, 0, stream>>>(sums, KCENT * DH);
    k_zero<<<dim3(KCENT / 256 + 1), blk, 0, stream>>>(cnt, KCENT);
    k_segsum<<<dim3(BHN * SQ / 8), blk, 0, stream>>>(qh, asg, sums, cnt);
    k_cent_update<<<dim3(KCENT * DH / 256), blk, 0, stream>>>(sums, cnt, cent, centH);
    k_cnorm<<<dim3(KCENT / 8), blk, 0, stream>>>(cent, cnorm);
  }
  // 6) assign keys to centroids, score, exact stable top-k, gather
  k_assign<<<dim3(BHN * SK / 128), blk, 0, stream>>>(kh, centH, cnorm, kc);
  k_dist<<<dim3(BHN * SK / 8), blk, 0, stream>>>(kh, kc, cent, dist);
  k_topk<<<dim3(BHN), blk, 0, stream>>>(dist, sel);
  k_gather<<<dim3(BHN * TOPK / 256), blk, 0, stream>>>(kh, vh, sel, ksel, vsel);
  // 7) flash attention over selected keys
  k_attn<<<dim3(BHN * (SQ / 128)), blk, 0, stream>>>(qh, ksel, vsel, attnH);
  // 8) output projection (WMMA GEMM) + final LN + residual
  k_gemm_nt<<<dim3((BATCH * SQ / 64) * (CDIM / 32)), blk, 0, stream>>>(
      attnH, woutH, Y3, BATCH * SQ, CDIM, INNER);
  k_final<<<dim3(BATCH * SQ / 8), blk, 0, stream>>>(Y3, on_g, on_b, gamma, qsrc, out);
}
